// IMRAM_81449759801702
// MI455X (gfx1250) — matrix operand order repaired and checked by reading, not compiled, not run on hardware
//
#include <hip/hip_runtime.h>

// ---------------------------------------------------------------------------
// IMRAM forward for MI455X (gfx1250, wave32, WMMA bf16 16x16x32).
// B=32, R=36, L=40, D=1024.  Query rows padded to 48, context rows to 64.
// ---------------------------------------------------------------------------

#define BB      32
#define RR      36
#define LL      40
#define DD      1024
#define QP      48          // padded query rows (40 or 36 -> 48)
#define CTXP    64          // padded context rows (36 or 40 -> 64, K-pad for GEMM2)
#define NPAIR   (BB * BB)   // 1024
#define NROWS   (NPAIR * QP)// 49152 rows in q / wc / qtmp buffers

typedef __attribute__((ext_vector_type(16))) __bf16 v16bf;
typedef __attribute__((ext_vector_type(8)))  float  v8f;

// ---- WMMA fragment loaders (wave32; A/B use contiguous k order, k = 16*(lane/16) + i) ----------

// A fragment, logical A[m][k] (16x32), source row-major with leading dim ld.
// lanes 0-15: M=lane, K = 0..15 ; lanes 16-31: M=lane-16, K = 16..31
__device__ __forceinline__ v16bf load_a_rm(const __bf16* src, int ld) {
  int lane = threadIdx.x & 31;
  int m  = lane & 15;
  int ko = (lane >> 4) << 4;
  v16bf a;
#pragma unroll
  for (int i = 0; i < 8; ++i) {
    a[i]     = src[m * ld + ko + i];
    a[i + 8] = src[m * ld + 8 + ko + i];
  }
  return a;
}

// A fragment from transposed source: A[m][k] = src[k*ld + m]
__device__ __forceinline__ v16bf load_a_tr(const __bf16* src, int ld) {
  int lane = threadIdx.x & 31;
  int m  = lane & 15;
  int ko = (lane >> 4) << 4;
  v16bf a;
#pragma unroll
  for (int i = 0; i < 8; ++i) {
    a[i]     = src[(ko + i) * ld + m];
    a[i + 8] = src[(8 + ko + i) * ld + m];
  }
  return a;
}

// B fragment, logical B[k][n] (32x16), source row-major [K][N] with ld.
// lanes 0-15: N=lane, K=0..15 ; lanes 16-31: N=lane-16, K=16..31
__device__ __forceinline__ v16bf load_b_rm(const __bf16* src, int ld) {
  int lane = threadIdx.x & 31;
  int n  = lane & 15;
  int kb = (lane >> 4) << 4;
  v16bf b;
#pragma unroll
  for (int i = 0; i < 16; ++i) b[i] = src[(kb + i) * ld + n];
  return b;
}

// B fragment from transposed source (row-major [N][K]): B[k][n] = src[n*ld + k]
__device__ __forceinline__ v16bf load_b_tr(const __bf16* src, int ld) {
  int lane = threadIdx.x & 31;
  int n  = lane & 15;
  int kb = (lane >> 4) << 4;
  v16bf b;
#pragma unroll
  for (int i = 0; i < 16; ++i) b[i] = src[n * ld + kb + i];
  return b;
}

// D/C fragment store (16x16 f32): VGPR i holds M=i (lanes 0-15) / M=8+i (16-31)
__device__ __forceinline__ void store_d_f32(float* dst, int ld, v8f d) {
  int lane = threadIdx.x & 31;
  int n  = lane & 15;
  int mb = (lane >> 4) << 3;
#pragma unroll
  for (int i = 0; i < 8; ++i) dst[(mb + i) * ld + n] = d[i];
}

__device__ __forceinline__ void store_d_bf16(__bf16* dst, int ld, v8f d) {
  int lane = threadIdx.x & 31;
  int n  = lane & 15;
  int mb = (lane >> 4) << 3;
#pragma unroll
  for (int i = 0; i < 8; ++i) dst[(mb + i) * ld + n] = (__bf16)d[i];
}

#define WMMA_BF16(A, Bm, C) \
  __builtin_amdgcn_wmma_f32_16x16x32_bf16(false, (A), false, (Bm), (short)0, (C), false, false)

// ---------------------------------------------------------------------------
// fp32 -> bf16 conversion with zero row padding.
// src: nmat matrices of [rv][cols]; dst: nmat matrices of [rp][cols].
__global__ __launch_bounds__(256)
void cvt_pad(const float* __restrict__ src, __bf16* __restrict__ dst,
             int nmat, int rv, int rp, int cols) {
  long total = (long)nmat * rp * cols;
  long stride = (long)gridDim.x * blockDim.x;
  for (long idx = (long)blockIdx.x * blockDim.x + threadIdx.x; idx < total; idx += stride) {
    int  c = (int)(idx % cols);
    long t = idx / cols;
    int  r = (int)(t % rp);
    long m = t / rp;
    float v = (r < rv) ? src[(m * rv + r) * cols + c] : 0.0f;
    dst[idx] = (__bf16)v;
  }
}

// ---------------------------------------------------------------------------
// Duplicate-caption mask + zero the output accumulator.
__global__ __launch_bounds__(256)
void dup_total_init(const float* __restrict__ cap, int* __restrict__ dup,
                    float* __restrict__ total) {
  int c = blockIdx.x, i = blockIdx.y;
  const int N = LL * DD;
  float s = 0.0f;
  for (int t = threadIdx.x; t < N; t += 256) {
    float d = cap[(long)c * N + t] - cap[(long)i * N + t];
    s += d * d;
  }
  __shared__ float red[256];
  red[threadIdx.x] = s;
  __syncthreads();
  for (int off = 128; off > 0; off >>= 1) {
    if (threadIdx.x < off) red[threadIdx.x] += red[threadIdx.x + off];
    __syncthreads();
  }
  if (threadIdx.x == 0) {
    dup[c * BB + i]   = (red[0] <= 1e-6f && c != i) ? 1 : 0;
    total[c * BB + i] = 0.0f;
  }
}

// ---------------------------------------------------------------------------
// q[(c*32+i)][l][:] = src[(sel)][l][:]   (sel = c for t2i, i for i2t)
__global__ __launch_bounds__(256)
void init_q(const __bf16* __restrict__ src, __bf16* __restrict__ q, int sel_is_c) {
  long total = (long)NPAIR * QP * DD;
  long stride = (long)gridDim.x * blockDim.x;
  for (long idx = (long)blockIdx.x * blockDim.x + threadIdx.x; idx < total; idx += stride) {
    int  d = (int)(idx % DD);
    long t = idx / DD;
    int  l = (int)(t % QP);
    int  pair = (int)(t / QP);
    int  c = pair >> 5, i = pair & 31;
    int  sel = sel_is_c ? c : i;
    q[idx] = src[((long)sel * CTXP + l) * DD + d];
  }
}

// ---------------------------------------------------------------------------
// Per-pair attention step:
//   S[ctx_row][q_row] = <ctx_row, q_row>                 (WMMA, K=1024)
//   leaky-relu + l2norm across q_rows, softmax*9 across ctx_rows
//   wc[q_row][:] = sum_ctx attn * ctx                    (WMMA, K=64)
__global__ __launch_bounds__(128)
void attn_step(const __bf16* __restrict__ qbuf,
               const __bf16* __restrict__ ctx,   // [32][CTXP][D], zero padded
               __bf16* __restrict__ wcbuf,       // [NPAIR][QP][D]
               int ctx_is_img, int Cv, int Qv) {
  int pair = blockIdx.x;
  int c = pair >> 5, i = pair & 31;
  int sel = ctx_is_img ? i : c;
  const __bf16* ctxm = ctx  + (long)sel  * CTXP * DD;
  const __bf16* qm   = qbuf + (long)pair * QP * DD;
  __bf16*       wcm  = wcbuf + (long)pair * QP * DD;

  __shared__ float  Sf[CTXP][QP];  // raw / normalized scores (rows 48..63 unused)
  __shared__ __bf16 Sb[CTXP][QP];  // softmaxed attention, bf16, K-padded with zeros

  int wave = threadIdx.x >> 5;

  // ---- GEMM1: 3x3 tiles of 16x16, K = D ----
  for (int t = wave; t < 9; t += 4) {
    int ct = t / 3, qt = t % 3;
    v8f acc = {};
    for (int k = 0; k < DD; k += 32) {
      v16bf a = load_a_rm(ctxm + (long)(ct * 16) * DD + k, DD);
      v16bf b = load_b_tr(qm   + (long)(qt * 16) * DD + k, DD);
      acc = WMMA_BF16(a, b, acc);
    }
    store_d_f32(&Sf[ct * 16][qt * 16], QP, acc);
  }
  __syncthreads();

  // ---- leaky-relu + l2norm across columns (query axis), per ctx row ----
  for (int r = threadIdx.x; r < QP; r += 128) {
    float ss = 0.0f;
    for (int q = 0; q < QP; ++q) {
      float x = Sf[r][q];
      x = x > 0.0f ? x : 0.1f * x;
      ss += x * x;
    }
    float inv = 1.0f / (sqrtf(ss) + 1e-8f);
    for (int q = 0; q < QP; ++q) {
      float x = Sf[r][q];
      x = x > 0.0f ? x : 0.1f * x;
      Sf[r][q] = x * inv;
    }
  }
  __syncthreads();

  // ---- softmax(9*S) across ctx rows, per query column; zero padded cols ----
  for (int q = threadIdx.x; q < QP; q += 128) {
    if (q < Qv) {
      float mx = -1e30f;
      for (int r = 0; r < Cv; ++r) mx = fmaxf(mx, 9.0f * Sf[r][q]);
      float sum = 0.0f;
      for (int r = 0; r < Cv; ++r) sum += __expf(9.0f * Sf[r][q] - mx);
      float inv = 1.0f / sum;
      for (int r = 0; r < CTXP; ++r)
        Sb[r][q] = (__bf16)((r < Cv) ? __expf(9.0f * Sf[r][q] - mx) * inv : 0.0f);
    } else {
      for (int r = 0; r < CTXP; ++r) Sb[r][q] = (__bf16)0.0f;
    }
  }
  __syncthreads();

  // ---- GEMM2: wc (QPxD) = attn^T (QPxCTXP) x ctx (CTXPxD), K = 64 ----
  for (int t = wave; t < 3 * (DD / 16); t += 4) {
    int qt = t >> 6;        // / 64
    int dt = t & 63;
    v8f acc = {};
#pragma unroll
    for (int k = 0; k < CTXP; k += 32) {
      v16bf a = load_a_tr(&Sb[k][qt * 16], QP);
      v16bf b = load_b_rm(ctxm + (long)k * DD + dt * 16, DD);
      acc = WMMA_BF16(a, b, acc);
    }
    store_d_bf16(wcm + (long)(qt * 16) * DD + dt * 16, DD, acc);
  }
}

// ---------------------------------------------------------------------------
// sim = mean_l cos(orig[sel][l], wc[pair][l]); total += dup ? -1 : sim
__global__ __launch_bounds__(256)
void sim_step(const float* __restrict__ orig,     // fp32, [32][Qv][D] unpadded
              const __bf16* __restrict__ wcbuf,
              const int* __restrict__ dup,
              float* __restrict__ total,
              int orig_is_cap, int Qv) {
  int pair = blockIdx.x;
  int c = pair >> 5, i = pair & 31;
  int sel = orig_is_cap ? c : i;
  const float*  om = orig  + (long)sel  * Qv * DD;
  const __bf16* wm = wcbuf + (long)pair * QP * DD;

  __shared__ float acc;
  if (threadIdx.x == 0) acc = 0.0f;
  __syncthreads();

  int wave = threadIdx.x >> 5, lane = threadIdx.x & 31;
  for (int l = wave; l < Qv; l += 8) {
    float dot = 0.0f, na = 0.0f, nw = 0.0f;
    for (int d = lane; d < DD; d += 32) {
      float a = om[(long)l * DD + d];
      float w = (float)wm[(long)l * DD + d];
      dot += a * w; na += a * a; nw += w * w;
    }
    for (int off = 16; off > 0; off >>= 1) {
      dot += __shfl_xor(dot, off, 32);
      na  += __shfl_xor(na,  off, 32);
      nw  += __shfl_xor(nw,  off, 32);
    }
    if (lane == 0) atomicAdd(&acc, dot / fmaxf(sqrtf(na) * sqrtf(nw), 1e-8f));
  }
  __syncthreads();
  if (threadIdx.x == 0)
    total[pair] += dup[pair] ? -1.0f : (acc / (float)Qv);
}

// ---------------------------------------------------------------------------
// Gated fusion: lin = tanh([q|wc] @ Wl^T + bl), gate = sigmoid([q|wc] @ Wg^T + bg)
// qtmp = q*gate + lin*(1-gate).     grid: (NROWS/64, D/64), 4 waves/block.
__global__ __launch_bounds__(128)
void gated_step(const __bf16* __restrict__ qbuf,
                const __bf16* __restrict__ wcbuf,
                const __bf16* __restrict__ Wl,   // [D][2D] bf16, row-major
                const __bf16* __restrict__ Wg,
                const float* __restrict__ bl,
                const float* __restrict__ bg,
                __bf16* __restrict__ qtmp) {
  int wave  = threadIdx.x >> 5;
  int mbase = blockIdx.x * 64 + wave * 16;
  int nbase = blockIdx.y * 64;

  v8f accl[4] = {{}, {}, {}, {}};
  v8f accg[4] = {{}, {}, {}, {}};

  for (int k = 0; k < 2 * DD; k += 32) {
    const __bf16* asrc = (k < DD) ? (qbuf  + (long)mbase * DD + k)
                                  : (wcbuf + (long)mbase * DD + (k - DD));
    v16bf a = load_a_rm(asrc, DD);
#pragma unroll
    for (int j = 0; j < 4; ++j) {
      v16bf bfl = load_b_tr(Wl + (long)(nbase + j * 16) * (2 * DD) + k, 2 * DD);
      accl[j] = WMMA_BF16(a, bfl, accl[j]);
      v16bf bfg = load_b_tr(Wg + (long)(nbase + j * 16) * (2 * DD) + k, 2 * DD);
      accg[j] = WMMA_BF16(a, bfg, accg[j]);
    }
  }

  int lane = threadIdx.x & 31;
  int mb   = mbase + ((lane >> 4) << 3);
#pragma unroll
  for (int j = 0; j < 4; ++j) {
    int n = nbase + j * 16 + (lane & 15);
    float bln = bl[n], bgn = bg[n];
#pragma unroll
    for (int e = 0; e < 8; ++e) {
      long row = mb + e;
      float lin  = tanhf(accl[j][e] + bln);
      float gate = 1.0f / (1.0f + __expf(-(accg[j][e] + bgn)));
      float qo   = (float)qbuf[row * DD + n];
      qtmp[row * DD + n] = (__bf16)(qo * gate + lin * (1.0f - gate));
    }
  }
}

// ---------------------------------------------------------------------------
// q = x / (||x|| + eps) per row of 1024.
__global__ __launch_bounds__(256)
void l2norm_rows(const __bf16* __restrict__ src, __bf16* __restrict__ dst) {
  long row = blockIdx.x;
  const __bf16* s = src + row * DD;
  float ss = 0.0f;
  for (int d = threadIdx.x; d < DD; d += 256) {
    float v = (float)s[d];
    ss += v * v;
  }
  __shared__ float red[256];
  red[threadIdx.x] = ss;
  __syncthreads();
  for (int off = 128; off > 0; off >>= 1) {
    if (threadIdx.x < off) red[threadIdx.x] += red[threadIdx.x + off];
    __syncthreads();
  }
  float inv = 1.0f / (sqrtf(red[0]) + 1e-8f);
  for (int d = threadIdx.x; d < DD; d += 256)
    dst[row * DD + d] = (__bf16)((float)s[d] * inv);
}

// ---------------------------------------------------------------------------
extern "C" void kernel_launch(void* const* d_in, const int* in_sizes, int n_in,
                              void* d_out, int out_size, void* d_ws, size_t ws_size,
                              hipStream_t stream) {
  const float* img    = (const float*)d_in[0];
  const float* cap    = (const float*)d_in[1];
  const float* W_lt2i = (const float*)d_in[2];
  const float* b_lt2i = (const float*)d_in[3];
  const float* W_gt2i = (const float*)d_in[4];
  const float* b_gt2i = (const float*)d_in[5];
  const float* W_li2t = (const float*)d_in[6];
  const float* b_li2t = (const float*)d_in[7];
  const float* W_gi2t = (const float*)d_in[8];
  const float* b_gi2t = (const float*)d_in[9];
  float* total = (float*)d_out;

  // ---- workspace carve-up (bf16 everywhere; ~312 MiB total) ----
  char* ws = (char*)d_ws;
  size_t off = 0;
  auto carve = [&](size_t bytes) -> char* {
    char* p = ws + off;
    off = (off + bytes + 255) & ~(size_t)255;
    return p;
  };
  __bf16* img_bf = (__bf16*)carve((size_t)BB * CTXP * DD * 2);
  __bf16* cap_bf = (__bf16*)carve((size_t)BB * CTXP * DD * 2);
  __bf16* Wl_t2i = (__bf16*)carve((size_t)DD * 2 * DD * 2);
  __bf16* Wg_t2i = (__bf16*)carve((size_t)DD * 2 * DD * 2);
  __bf16* Wl_i2t = (__bf16*)carve((size_t)DD * 2 * DD * 2);
  __bf16* Wg_i2t = (__bf16*)carve((size_t)DD * 2 * DD * 2);
  int*    dup    = (int*)   carve((size_t)NPAIR * 4);
  __bf16* q_bf   = (__bf16*)carve((size_t)NROWS * DD * 2);
  __bf16* wc_bf  = (__bf16*)carve((size_t)NROWS * DD * 2);
  __bf16* qtmp   = (__bf16*)carve((size_t)NROWS * DD * 2);

  // ---- precompute: bf16 conversions + dup mask + zero output ----
  cvt_pad<<<1024, 256, 0, stream>>>(img, img_bf, BB, RR, CTXP, DD);
  cvt_pad<<<1024, 256, 0, stream>>>(cap, cap_bf, BB, LL, CTXP, DD);
  cvt_pad<<<2048, 256, 0, stream>>>(W_lt2i, Wl_t2i, 1, DD, DD, 2 * DD);
  cvt_pad<<<2048, 256, 0, stream>>>(W_gt2i, Wg_t2i, 1, DD, DD, 2 * DD);
  cvt_pad<<<2048, 256, 0, stream>>>(W_li2t, Wl_i2t, 1, DD, DD, 2 * DD);
  cvt_pad<<<2048, 256, 0, stream>>>(W_gi2t, Wg_i2t, 1, DD, DD, 2 * DD);
  {
    dim3 g(BB, BB);
    dup_total_init<<<g, 256, 0, stream>>>(cap, dup, total);
  }

  auto run_direction = [&](int t2i,
                           const __bf16* ctx, const float* orig, int Cv, int Qv,
                           const __bf16* Wl, const float* bl,
                           const __bf16* Wg, const float* bg) {
    // t2i: query = caption words (sel=c), context = image regions (sel=i)
    init_q<<<4096, 256, 0, stream>>>(t2i ? cap_bf : img_bf, q_bf, /*sel_is_c=*/t2i);
    for (int s = 0; s < 3; ++s) {
      attn_step<<<NPAIR, 128, 0, stream>>>(q_bf, ctx, wc_bf,
                                           /*ctx_is_img=*/t2i, Cv, Qv);
      sim_step<<<NPAIR, 256, 0, stream>>>(orig, wc_bf, dup, total,
                                          /*orig_is_cap=*/t2i, Qv);
      if (s < 2) {  // last-step q update is unused by the reference output
        dim3 g(NROWS / 64, DD / 64);
        gated_step<<<g, 128, 0, stream>>>(q_bf, wc_bf, Wl, Wg, bl, bg, qtmp);
        l2norm_rows<<<NROWS, 256, 0, stream>>>(qtmp, q_bf);
      }
    }
  };

  // text-to-image: ctx = image regions (Cv=R), queries = words (Qv=L)
  run_direction(1, img_bf, cap, RR, LL, Wl_t2i, b_lt2i, Wg_t2i, b_gt2i);
  // image-to-text: ctx = caption words (Cv=L), queries = regions (Qv=R)
  run_direction(0, cap_bf, img, LL, RR, Wl_i2t, b_li2t, Wg_i2t, b_gi2t);
}
